// STGCNEdgeWeightedUnit_7842610282873
// MI455X (gfx1250) — matrix operand order repaired and checked by reading, not compiled, not run on hardware
//
#include <hip/hip_runtime.h>

// Problem constants (from reference)
#define NB     64     // batch
#define CIN    64
#define COUT   64
#define TLEN   300
#define VV     25
#define KPART  3
#define TTILE  4      // timesteps per block (300 % 4 == 0)
#define MT     100    // TTILE*VV, m-positions per tile
#define NBLK_T 75     // TLEN / TTILE
#define CNT_PER_CH 480000.0f   // N*T*V
#define BN_EPS 1e-5f

typedef __attribute__((ext_vector_type(16))) __bf16 v16bf;
typedef __attribute__((ext_vector_type(8)))  __bf16 v8bf;
typedef __attribute__((ext_vector_type(8)))  float  v8f;

static __device__ __forceinline__ v16bf mk16(v8bf lo, v8bf hi) {
  return __builtin_shufflevector(lo, hi, 0,1,2,3,4,5,6,7,8,9,10,11,12,13,14,15);
}

// ws layout (floats): [0..63]=sum, [64..127]=sumsq, [128..191]=bsum
__global__ void prep_kernel(const float* __restrict__ b, float* __restrict__ ws) {
  int i = threadIdx.x;
  if (i < 128) ws[i] = 0.0f;
  if (i < 64)  ws[128 + i] = b[i] + b[64 + i] + b[128 + i];
}

__global__ __launch_bounds__(256) void stgcn_main(
    const float* __restrict__ x, const float* __restrict__ A,
    const float* __restrict__ Mmask, const float* __restrict__ W,
    float* __restrict__ out, float* __restrict__ ws) {
  // LDS: 24576 + 14336 + 16384 + 6144 + 768 B = ~61 KB
  __shared__ __align__(16) __bf16 Wb[KPART][64][64];   // W as bf16, [k][o][c]
  __shared__ __align__(16) __bf16 Xb[112][64];         // x tile transposed [m][c]
  __shared__ __align__(16) __bf16 Yb[256][32];         // [o*4+t][v], v padded to 32
  __shared__ __align__(16) __bf16 AwT[KPART][32][32];  // (A*M)^T: [k][w][v], zero-padded
  __shared__ float red[64];
  __shared__ float redsq[64];
  __shared__ float bsumS[64];

  const int tid  = threadIdx.x;
  const int lane = tid & 31;
  const int wv   = tid >> 5;        // 8 waves
  const int l15  = lane & 15;
  const int h    = lane >> 4;       // lane half

  const int n  = blockIdx.x / NBLK_T;
  const int tb = blockIdx.x % NBLK_T;
  const int t0 = tb * TTILE;

  // ---------------- preamble ----------------
  for (int i = tid; i < KPART*64*64; i += 256)
    ((__bf16*)Wb)[i] = (__bf16)W[i];

  { // x[n][c][t0*25 .. t0*25+99] is contiguous per c; write transposed
    const float* xb = x + (size_t)n * CIN * (TLEN*VV) + (size_t)t0 * VV;
    for (int i = tid; i < CIN * MT; i += 256) {
      int c = i / MT, m = i % MT;
      Xb[m][c] = (__bf16)xb[(size_t)c * (TLEN*VV) + m];
    }
  }
  for (int i = tid; i < 256*32; i += 256) ((__bf16*)Yb)[i] = (__bf16)0.0f;
  for (int i = tid; i < KPART*32*32; i += 256) ((__bf16*)AwT)[i] = (__bf16)0.0f;
  if (tid < 64) { red[tid] = 0.0f; redsq[tid] = 0.0f; bsumS[tid] = ws[128 + tid]; }
  __syncthreads();
  // fill AwT (transposed, after zero-pad is visible)
  for (int i = tid; i < KPART*VV*VV; i += 256) {
    int k = i / (VV*VV); int rem = i % (VV*VV);
    int v = rem / VV, w = rem % VV;
    AwT[k][w][v] = (__bf16)(A[i] * Mmask[i]);
  }
  __syncthreads();

  // per-wave persistent accumulators: 4 output tiles of 16x16 f32
  v8f acc[4];
  #pragma unroll
  for (int i = 0; i < 4; ++i) acc[i] = (v8f){0.f,0.f,0.f,0.f,0.f,0.f,0.f,0.f};

  for (int k = 0; k < KPART; ++k) {
    // ---- stage 1: Y = W_k (64x64) x Xtile (64x100) via bf16 WMMA ----
    for (int tile = wv; tile < 28; tile += 8) {     // 4 o-tiles x 7 m-tiles
      const int mt_o = tile & 3;
      const int nt_m = tile >> 2;
      const int orow = mt_o * 16 + l15;             // A-frag row
      const int mrow = nt_m * 16 + l15;             // B-frag col (m position)
      v8f c = (v8f){0.f,0.f,0.f,0.f,0.f,0.f,0.f,0.f};
      #pragma unroll
      for (int kk = 0; kk < 64; kk += 32) {
        v8bf alo = *(const v8bf*)&Wb[k][orow][kk + h*8];
        v8bf ahi = *(const v8bf*)&Wb[k][orow][kk + 16 + h*8];
        v8bf blo = *(const v8bf*)&Xb[mrow][kk + h*8];
        v8bf bhi = *(const v8bf*)&Xb[mrow][kk + 16 + h*8];
        c = __builtin_amdgcn_wmma_f32_16x16x32_bf16(
                false, mk16(alo, ahi), false, mk16(blo, bhi),
                (short)0, c, false, false);
      }
      if (mrow < MT) {                              // valid m only
        const int tt = mrow / VV, v = mrow % VV;
        #pragma unroll
        for (int j = 0; j < 8; ++j) {
          const int o = mt_o * 16 + 8*h + j;        // D row -> output channel
          Yb[o * TTILE + tt][v] = (__bf16)c[j];
        }
      }
    }
    __syncthreads();

    // ---- stage 2: out += Y (256x32,K=v) x AwT_k (32x32,N=w) ----
    #pragma unroll
    for (int i = 0; i < 4; ++i) {
      const int tile2 = wv * 4 + i;                 // 32 tiles total
      const int mt_r  = tile2 >> 1;
      const int nt_w  = tile2 & 1;
      const int r     = mt_r * 16 + l15;
      const int wcol  = nt_w * 16 + l15;            // >=25 rows of AwT are zero
      v8bf alo = *(const v8bf*)&Yb[r][h*8];
      v8bf ahi = *(const v8bf*)&Yb[r][16 + h*8];
      v8bf blo = *(const v8bf*)&AwT[k][wcol][h*8];
      v8bf bhi = *(const v8bf*)&AwT[k][wcol][16 + h*8];
      acc[i] = __builtin_amdgcn_wmma_f32_16x16x32_bf16(
                   false, mk16(alo, ahi), false, mk16(blo, bhi),
                   (short)0, acc[i], false, false);
    }
    __syncthreads();  // Yb safe to overwrite next k
  }

  // ---------------- epilogue: bias add, store pre-BN, BN stats ----------------
  float ls[4]  = {0.f, 0.f, 0.f, 0.f};
  float lss[4] = {0.f, 0.f, 0.f, 0.f};
  #pragma unroll
  for (int i = 0; i < 4; ++i) {
    const int tile2 = wv * 4 + i;
    const int mt_r  = tile2 >> 1;
    const int nt_w  = tile2 & 1;
    const int w     = nt_w * 16 + l15;
    if (w < VV) {
      #pragma unroll
      for (int j = 0; j < 8; ++j) {
        const int r = mt_r * 16 + 8*h + j;
        const int o = r >> 2;
        const int t = t0 + (r & 3);
        const float val = acc[i][j] + bsumS[o];
        out[((size_t)(n * 64 + o) * TLEN + t) * VV + w] = val;
        const int p = ((i >> 1) << 1) | (j >> 2);
        ls[p]  += val;
        lss[p] += val * val;
      }
    }
  }
  #pragma unroll
  for (int p = 0; p < 4; ++p) {
    const int o = 8*wv + 4*(p >> 1) + 2*h + (p & 1);
    atomicAdd(&red[o], ls[p]);
    atomicAdd(&redsq[o], lss[p]);
  }
  __syncthreads();
  if (tid < 64) {
    atomicAdd(&ws[tid], red[tid]);
    atomicAdd(&ws[64 + tid], redsq[tid]);
  }
}

__global__ void finalize_kernel(float* __restrict__ out, const float* __restrict__ ws,
                                const float* __restrict__ gamma,
                                const float* __restrict__ beta) {
  const float invc = 1.0f / CNT_PER_CH;
  const int total4 = NB * COUT * TLEN * VV / 4;   // 7,680,000 float4s
  const int stride = gridDim.x * blockDim.x;
  for (int i = blockIdx.x * blockDim.x + threadIdx.x; i < total4; i += stride) {
    const int o = (i / 1875) & 63;                // 7500 floats per channel row-block
    const float mean = ws[o] * invc;
    const float var  = ws[64 + o] * invc - mean * mean;
    const float sc = gamma[o] * rsqrtf(var + BN_EPS);
    const float sh = beta[o] - mean * sc;
    float4 v = reinterpret_cast<float4*>(out)[i];
    v.x = fmaxf(fmaf(v.x, sc, sh), 0.0f);
    v.y = fmaxf(fmaf(v.y, sc, sh), 0.0f);
    v.z = fmaxf(fmaf(v.z, sc, sh), 0.0f);
    v.w = fmaxf(fmaf(v.w, sc, sh), 0.0f);
    reinterpret_cast<float4*>(out)[i] = v;
  }
}

extern "C" void kernel_launch(void* const* d_in, const int* in_sizes, int n_in,
                              void* d_out, int out_size, void* d_ws, size_t ws_size,
                              hipStream_t stream) {
  const float* x     = (const float*)d_in[0];
  const float* A     = (const float*)d_in[1];
  const float* Mm    = (const float*)d_in[2];
  const float* W     = (const float*)d_in[3];
  const float* b     = (const float*)d_in[4];
  const float* gamma = (const float*)d_in[5];
  const float* beta  = (const float*)d_in[6];
  float* out = (float*)d_out;
  float* ws  = (float*)d_ws;

  prep_kernel<<<1, 256, 0, stream>>>(b, ws);
  stgcn_main<<<dim3(NB * NBLK_T), dim3(256), 0, stream>>>(x, A, Mm, W, out, ws);
  finalize_kernel<<<2048, 256, 0, stream>>>(out, ws, gamma, beta);
}
